// EGNNLayer_50491635531835
// MI455X (gfx1250) — matrix operand order repaired and checked by reading, not compiled, not run on hardware
//
#include <hip/hip_runtime.h>
#include <hip/hip_bf16.h>

// ---------------------------------------------------------------------------
// EGNN layer for MI455X (gfx1250), fused bf16-WMMA implementation.
// B=2, N=512, H=128.  Dominant work: two (B*N^2)x128x128 GEMMs fused per
// 16(i) x 1(j) tile, never materializing m_ij in HBM.
// Round 3: attention logits computed by an extra WMMA (aw replicated across
// the 16 B-columns) instead of ds_bpermute shuffle chains; att is a per-row
// scalar so it commutes through GEMM2: silu((m~*att)@cw1+cb1)
//   == silu(att*(m~@cw1)+cb1).  Removes attLDS + one barrier + 32 bpermutes
// per j-step; wmma per j-step: 16 -> 20.
// ---------------------------------------------------------------------------

#define Hc 128
#define Nc 512
#define Bc 2
#define JS 8                  // j-splits per i-tile (parallelism + partials)
#define JCH (Nc / JS)         // 64 j per block
#define NT (Nc / 16)          // 32 i-tiles per batch
#define NTILES (Bc * NT)      // 64

typedef __attribute__((ext_vector_type(16))) __bf16 v16bf;
typedef __attribute__((ext_vector_type(8)))  float  v8f;
typedef __attribute__((ext_vector_type(2)))  __bf16 bf2;

// fast sigmoid/silu: v_exp_f32 + v_rcp_f32 (both TRANS ops, co-execute w/ WMMA)
__device__ __forceinline__ float fsigm(float v) {
  return __builtin_amdgcn_rcpf(1.f + __expf(-v));
}
__device__ __forceinline__ float fsilu(float v) { return v * fsigm(v); }

// sum across the 16 lanes of each wave32 half (xor masks < 16 stay in-half)
__device__ __forceinline__ float rowsum16(float v) {
  v += __shfl_xor(v, 1, 32);
  v += __shfl_xor(v, 2, 32);
  v += __shfl_xor(v, 4, 32);
  v += __shfl_xor(v, 8, 32);
  return v;
}

// ---------------------------------------------------------------------------
// Kernel 1: hi_proj = h @ ew1[0:128], hj_proj = h @ ew1[128:256]
// one block per node, 128 threads (negligible cost: ~67 MFLOP)
// ---------------------------------------------------------------------------
__global__ __launch_bounds__(Hc) void egnn_proj(
    const float* __restrict__ h, const float* __restrict__ ew1,
    float* __restrict__ hiP, float* __restrict__ hjP) {
  const int node = blockIdx.x;            // b*N + i
  const int t = threadIdx.x;
  __shared__ float hrow[Hc];
  hrow[t] = h[(size_t)node * Hc + t];
  __syncthreads();
  float si = 0.f, sj = 0.f;
  #pragma unroll 8
  for (int k = 0; k < Hc; ++k) {
    const float hv = hrow[k];
    si += hv * ew1[k * Hc + t];
    sj += hv * ew1[(Hc + k) * Hc + t];
  }
  hiP[(size_t)node * Hc + t] = si;
  hjP[(size_t)node * Hc + t] = sj;
}

// ---------------------------------------------------------------------------
// Kernel 2: fused edge kernel.
// Block = 128 threads = 4 waves.  Block handles (b, 16 i-rows, 64 j's).
// Wave w owns output columns [32w, 32w+32) for both 128x128 GEMMs.
// ---------------------------------------------------------------------------
__global__ __launch_bounds__(128) void egnn_edge(
    const float* __restrict__ x,
    const float* __restrict__ ew1, const float* __restrict__ eb1,
    const float* __restrict__ ew2, const float* __restrict__ eb2,
    const float* __restrict__ cw1, const float* __restrict__ cb1,
    const float* __restrict__ cw2, const float* __restrict__ aw,
    const float* __restrict__ ab,
    const float* __restrict__ hiP, const float* __restrict__ hjP,
    float* __restrict__ ws_mi, float* __restrict__ ws_cd) {
  const int blk  = blockIdx.x;
  const int js   = blk & (JS - 1);
  const int tile = blk / JS;              // b*NT + it
  const int b    = tile / NT;
  const int i0   = (tile % NT) << 4;
  const int tid  = threadIdx.x;
  const int wave = tid >> 5;
  const int lane = tid & 31;
  const int lo16 = lane & 15;
  const int half = lane >> 4;
  const int nb0  = wave * 32;             // this wave's first output column

  __shared__ __align__(16) __bf16 abuf[16 * Hc];   // A = silu(pre), bf16
  __shared__ __align__(16) __bf16 mbuf[16 * Hc];   // m~ (pre-att), bf16
  __shared__ __align__(16) float  cwLDS[64];       // [row][wave] partial dots

  // ---- cooperative A-build mapping: thread -> (row ar, 16 k's at ak) ----
  const int ar = tid >> 3;                // 0..15
  const int ak = (tid & 7) << 4;          // 0,16,...,112
  float preV[16], wrV[16];
  {
    const float* hiRow = hiP + (size_t)(b * Nc + i0 + ar) * Hc + ak;
    const float* wr = ew1 + 2 * Hc * Hc;  // row 256 of ew1 = w_r
    #pragma unroll
    for (int q = 0; q < 16; ++q) {
      preV[q] = hiRow[q] + eb1[ak + q];   // fold eb1 into the i-projection
      wrV[q]  = wr[ak + q];
    }
  }
  const float xi0 = x[(size_t)(b * Nc + i0 + ar) * 3 + 0];
  const float xi1 = x[(size_t)(b * Nc + i0 + ar) * 3 + 1];
  const float xi2 = x[(size_t)(b * Nc + i0 + ar) * 3 + 2];

  // ---- resident bf16 B fragments ----
  // B frags use contiguous k (same as A): VGPR i, lanes<16: K=2i,2i+1; lanes>=16: K=16+2i,16+2i+1.
  // B1 = ew2 (GEMM1), B2 = cw1 (GEMM2), AwB = aw broadcast to all 16 columns
  // (so the att-WMMA leaves the row dot-product in every lane of that row).
  v16bf B1[4][2], B2[4][2], AwB[4];
  #pragma unroll
  for (int f = 0; f < 4; ++f) {
    #pragma unroll
    for (int t = 0; t < 2; ++t) {
      const int n = nb0 + t * 16 + lo16;
      #pragma unroll
      for (int i = 0; i < 8; ++i) {
        const int kk = 32 * f + 16 * half + 2 * i;
        B1[f][t][2 * i]     = (__bf16)ew2[kk * Hc + n];
        B1[f][t][2 * i + 1] = (__bf16)ew2[(kk + 1) * Hc + n];
        B2[f][t][2 * i]     = (__bf16)cw1[kk * Hc + n];
        B2[f][t][2 * i + 1] = (__bf16)cw1[(kk + 1) * Hc + n];
      }
    }
    #pragma unroll
    for (int i = 0; i < 8; ++i) {
      const int kk = 32 * f + 16 * half + 2 * i;
      AwB[f][2 * i]     = (__bf16)aw[kk];       // same for every column
      AwB[f][2 * i + 1] = (__bf16)aw[kk + 1];
    }
  }
  float eb2v[2], cb1v[2], cw2v[2];
  #pragma unroll
  for (int t = 0; t < 2; ++t) {
    const int n = nb0 + t * 16 + lo16;
    eb2v[t] = eb2[n]; cb1v[t] = cb1[n]; cw2v[t] = cw2[n];
  }
  const float ab0 = ab[0];

  float macc[2][8];                       // m_i accumulator (C-layout rows)
  #pragma unroll
  for (int t = 0; t < 2; ++t)
    #pragma unroll
    for (int v = 0; v < 8; ++v) macc[t][v] = 0.f;

  const bool rowLead = ((tid & 7) == 0);  // one thread per i-row (row = ar)
  float cacc0 = 0.f, cacc1 = 0.f, cacc2 = 0.f;

  for (int jj = 0; jj < JCH; ++jj) {
    const int j = js * JCH + jj;
    const float xj0 = x[(size_t)(b * Nc + j) * 3 + 0];
    const float xj1 = x[(size_t)(b * Nc + j) * 3 + 1];
    const float xj2 = x[(size_t)(b * Nc + j) * 3 + 2];
    const float d0 = xi0 - xj0, d1 = xi1 - xj1, d2 = xi2 - xj2;
    const float rad = d0 * d0 + d1 * d1 + d2 * d2;

    const float* hjRow = hjP + (size_t)(b * Nc + j) * Hc + ak;
    if (jj + 1 < JCH)                     // global_prefetch_b8 next j row
      __builtin_prefetch(hjP + (size_t)(b * Nc + j + 1) * Hc + ak, 0, 0);

    // ---- build A = silu(pre) into LDS (bf16, row-major 16x128) ----
    #pragma unroll
    for (int q = 0; q < 16; q += 2) {
      const float p0 = preV[q]     + hjRow[q]     + rad * wrV[q];
      const float p1 = preV[q + 1] + hjRow[q + 1] + rad * wrV[q + 1];
      bf2 pk;
      pk[0] = (__bf16)fsilu(p0);
      pk[1] = (__bf16)fsilu(p1);
      *(bf2*)&abuf[ar * Hc + ak + q] = pk;
    }
    __syncthreads();  // B0: abuf ready

    // ---- GEMM1: m~ = silu( A @ ew2 + eb2 ) for this wave's 32 cols ----
    v8f acc[2];
    #pragma unroll
    for (int t = 0; t < 2; ++t)
      #pragma unroll
      for (int q = 0; q < 8; ++q) acc[t][q] = eb2v[t];
    #pragma unroll
    for (int f = 0; f < 4; ++f) {
      v16bf a;  // A 16x32 bf16 fragment: VGPR i: K=2i(+16*half)
      #pragma unroll
      for (int i = 0; i < 8; ++i) {
        const int k0 = 32 * f + 2 * i + half * 16;
        const bf2 pk = *(const bf2*)&abuf[lo16 * Hc + k0];
        a[2 * i] = pk[0]; a[2 * i + 1] = pk[1];
      }
      acc[0] = __builtin_amdgcn_wmma_f32_16x16x32_bf16(false, a, false, B1[f][0], (short)0, acc[0], false, false);
      acc[1] = __builtin_amdgcn_wmma_f32_16x16x32_bf16(false, a, false, B1[f][1], (short)0, acc[1], false, false);
    }
    float mt[2][8];
    #pragma unroll
    for (int t = 0; t < 2; ++t)
      #pragma unroll
      for (int v = 0; v < 8; ++v) mt[t][v] = fsilu(acc[t][v]);

    // ---- stage pre-att m~ (bf16) for the att-WMMA and GEMM2 ----
    #pragma unroll
    for (int v = 0; v < 8; ++v) {
      const int row = v + half * 8;
      #pragma unroll
      for (int t = 0; t < 2; ++t)
        mbuf[row * Hc + nb0 + t * 16 + lo16] = (__bf16)mt[t][v];
    }
    __syncthreads();  // B1: mbuf ready

    // ---- load A' = m~ fragments once; reuse for att-WMMA + GEMM2 ----
    v16bf a2[4];
    #pragma unroll
    for (int f = 0; f < 4; ++f)
      #pragma unroll
      for (int i = 0; i < 8; ++i) {
        const int k0 = 32 * f + 2 * i + half * 16;
        const bf2 pk = *(const bf2*)&mbuf[lo16 * Hc + k0];
        a2[f][2 * i] = pk[0]; a2[f][2 * i + 1] = pk[1];
      }

    // ---- attention: att[row] = sigmoid(m~[row,:] . aw + ab) via WMMA ----
    v8f accA;
    #pragma unroll
    for (int q = 0; q < 8; ++q) accA[q] = ab0;
    #pragma unroll
    for (int f = 0; f < 4; ++f)
      accA = __builtin_amdgcn_wmma_f32_16x16x32_bf16(false, a2[f], false, AwB[f], (short)0, accA, false, false);
    float att[8];
    #pragma unroll
    for (int v = 0; v < 8; ++v) att[v] = fsigm(accA[v]);

    // ---- m_i accumulation: m_ij = m~ * att (f32, in registers) ----
    #pragma unroll
    for (int t = 0; t < 2; ++t)
      #pragma unroll
      for (int v = 0; v < 8; ++v) macc[t][v] += mt[t][v] * att[v];

    // ---- GEMM2: G = m~ @ cw1; cw = silu(att*G + cb1) @ cw2 ----
    v8f acc2[2];
    #pragma unroll
    for (int t = 0; t < 2; ++t)
      #pragma unroll
      for (int q = 0; q < 8; ++q) acc2[t][q] = 0.f;
    #pragma unroll
    for (int f = 0; f < 4; ++f) {
      acc2[0] = __builtin_amdgcn_wmma_f32_16x16x32_bf16(false, a2[f], false, B2[f][0], (short)0, acc2[0], false, false);
      acc2[1] = __builtin_amdgcn_wmma_f32_16x16x32_bf16(false, a2[f], false, B2[f][1], (short)0, acc2[1], false, false);
    }
    #pragma unroll
    for (int v = 0; v < 8; ++v) {
      float p = fsilu(att[v] * acc2[0][v] + cb1v[0]) * cw2v[0]
              + fsilu(att[v] * acc2[1][v] + cb1v[1]) * cw2v[1];
      p = rowsum16(p);
      if (lane == 0)  cwLDS[v * 4 + wave] = p;
      if (lane == 16) cwLDS[(v + 8) * 4 + wave] = p;
    }
    __syncthreads();  // B2: cwLDS ready

    // ---- coord update accumulation (diag auto-zero: coord_diff[i,i]=0) ----
    if (rowLead) {
      const float4 c4 = *(const float4*)&cwLDS[ar * 4];
      const float cw = c4.x + c4.y + c4.z + c4.w;
      cacc0 += (xi0 - xj0) * cw;
      cacc1 += (xi1 - xj1) * cw;
      cacc2 += (xi2 - xj2) * cw;
    }
  }

  // ---- write per-(tile, j-split) partials ----
  #pragma unroll
  for (int t = 0; t < 2; ++t)
    #pragma unroll
    for (int v = 0; v < 8; ++v) {
      const int row = v + half * 8;
      const int col = nb0 + t * 16 + lo16;
      ws_mi[(size_t)blk * (16 * Hc) + row * Hc + col] = macc[t][v];
    }
  if (rowLead) {
    ws_cd[blk * 48 + ar * 3 + 0] = cacc0;
    ws_cd[blk * 48 + ar * 3 + 1] = cacc1;
    ws_cd[blk * 48 + ar * 3 + 2] = cacc2;
  }
}

// ---------------------------------------------------------------------------
// Kernel 3: reduce j-split partials, node MLP, outputs.
// ---------------------------------------------------------------------------
__global__ __launch_bounds__(Hc) void egnn_node(
    const float* __restrict__ h, const float* __restrict__ x,
    const float* __restrict__ nw1, const float* __restrict__ nb1,
    const float* __restrict__ nw2, const float* __restrict__ nb2,
    const float* __restrict__ ws_mi, const float* __restrict__ ws_cd,
    float* __restrict__ hout, float* __restrict__ xout) {
  const int node = blockIdx.x;            // b*N + i
  const int b = node / Nc;
  const int i = node % Nc;
  const int t = threadIdx.x;
  const int tile = b * NT + (i >> 4);
  const int r = i & 15;

  __shared__ float cat[2 * Hc];
  __shared__ float s1[Hc];

  float mi = 0.f;
  #pragma unroll
  for (int jsp = 0; jsp < JS; ++jsp)
    mi += ws_mi[(size_t)(tile * JS + jsp) * (16 * Hc) + r * Hc + t];
  cat[t] = h[(size_t)node * Hc + t];
  cat[Hc + t] = mi;
  __syncthreads();

  float y1 = nb1[t];
  #pragma unroll 8
  for (int k = 0; k < 2 * Hc; ++k) y1 += cat[k] * nw1[k * Hc + t];
  s1[t] = fsilu(y1);
  __syncthreads();

  float y2 = nb2[t];
  #pragma unroll 8
  for (int k = 0; k < Hc; ++k) y2 += s1[k] * nw2[k * Hc + t];
  hout[(size_t)node * Hc + t] = cat[t] + y2;

  if (t < 3) {
    float cd = 0.f;
    #pragma unroll
    for (int jsp = 0; jsp < JS; ++jsp)
      cd += ws_cd[(tile * JS + jsp) * 48 + r * 3 + t];
    xout[(size_t)node * 3 + t] = x[(size_t)node * 3 + t] + cd / (float)Nc;
  }
}

// ---------------------------------------------------------------------------
// Launch.  Workspace layout (floats):
//   hiP: B*N*H (131072) | hjP: 131072 | ws_mi: 64*JS*16*H (1048576)
//   | ws_cd: 64*JS*48 (24576)  => ~5.4 MB total.
// ---------------------------------------------------------------------------
extern "C" void kernel_launch(void* const* d_in, const int* in_sizes, int n_in,
                              void* d_out, int out_size, void* d_ws, size_t ws_size,
                              hipStream_t stream) {
  const float* h   = (const float*)d_in[0];
  const float* x   = (const float*)d_in[1];
  // d_in[2] = edge_index (unused by the reference math)
  const float* ew1 = (const float*)d_in[3];
  const float* eb1 = (const float*)d_in[4];
  const float* ew2 = (const float*)d_in[5];
  const float* eb2 = (const float*)d_in[6];
  const float* nw1 = (const float*)d_in[7];
  const float* nb1 = (const float*)d_in[8];
  const float* nw2 = (const float*)d_in[9];
  const float* nb2 = (const float*)d_in[10];
  const float* cw1 = (const float*)d_in[11];
  const float* cb1 = (const float*)d_in[12];
  const float* cw2 = (const float*)d_in[13];
  const float* aw  = (const float*)d_in[14];
  const float* ab  = (const float*)d_in[15];

  float* ws   = (float*)d_ws;
  float* hiP  = ws;
  float* hjP  = hiP + (size_t)Bc * Nc * Hc;
  float* wmi  = hjP + (size_t)Bc * Nc * Hc;
  float* wcd  = wmi + (size_t)NTILES * JS * 16 * Hc;

  float* hout = (float*)d_out;
  float* xout = hout + (size_t)Bc * Nc * Hc;

  egnn_proj<<<Bc * Nc, Hc, 0, stream>>>(h, ew1, hiP, hjP);
  egnn_edge<<<NTILES * JS, 128, 0, stream>>>(x, ew1, eb1, ew2, eb2, cw1, cb1,
                                             cw2, aw, ab, hiP, hjP, wmi, wcd);
  egnn_node<<<Bc * Nc, Hc, 0, stream>>>(h, x, nw1, nb1, nw2, nb2, wmi, wcd,
                                        hout, xout);
}
